// wave_cell_71777493450867
// MI455X (gfx1250) — matrix operand order repaired and checked by reading, not compiled, not run on hardware
//
#include <hip/hip_runtime.h>
#include <hip/hip_bf16.h>

// ---------------------------------------------------------------------------
// RedNet wave-cell for MI455X (gfx1250, wave32, WMMA).
// conv layers -> implicit GEMM via v_wmma_f32_16x16x32_f16 (f16 A/B, f32 acc).
// Activations: f16 NHWC (channel fastest, 32 ch = 64B per pixel) in d_ws.
// ---------------------------------------------------------------------------

typedef _Float16 v16h __attribute__((ext_vector_type(16)));
typedef _Float16 h8   __attribute__((ext_vector_type(8)));
typedef float    v8f  __attribute__((ext_vector_type(8)));

union BU { v16h v; h8 h[2]; };

constexpr int IMG    = 512;
constexpr int NBATCH = 8;
constexpr int HW     = IMG * IMG;
constexpr int NPX    = NBATCH * HW;       // 2,097,152
constexpr int CH     = 32;                // padded channel count (NHWC f16)
constexpr int TILE   = 32;                // output tile edge per workgroup
constexpr int HALO   = TILE + 2;          // 34 (3x3 SAME conv)
constexpr int PITCH  = 40;                // halfs per pixel in LDS (32 data + 8 pad)
                                          // 80B = 20 words: conflict-free stride mod 64 banks
constexpr int SMEM_BYTES = HALO * HALO * PITCH * (int)sizeof(_Float16); // 92,480

// ---------------------------------------------------------------------------
// Kernel 1: elementwise + Laplacian prep.
//  - V = x - H0 ; lap = conv(x, lapk) ; H = 2x - C0[0] + c2*lap
//  - writes d_out sections H, C, V
//  - builds rednet input act0 = Cv/sf as f16 NHWC, channels 4..31 zero-padded
// ---------------------------------------------------------------------------
__global__ void __launch_bounds__(256) prep_kernel(
                            const float* __restrict__ xin_p,
                            const float* __restrict__ H0,
                            const float* __restrict__ C0,
                            const float* __restrict__ c2,
                            const float* __restrict__ sf,
                            const float* __restrict__ lapk,
                            float* __restrict__ out,
                            _Float16* __restrict__ act) {
  int i = blockIdx.x * blockDim.x + threadIdx.x;
  if (i >= NPX) return;
  int b = i / HW, r = i % HW;
  int y = r / IMG, x = r % IMG;

  float xin = xin_p[i];
  float v   = xin - H0[i];

  float lap = 0.f;
#pragma unroll
  for (int ky = 0; ky < 3; ++ky)
#pragma unroll
    for (int kx = 0; kx < 3; ++kx) {
      int yy = y + ky - 1, xx = x + kx - 1;
      float s = (yy >= 0 && yy < IMG && xx >= 0 && xx < IMG)
                    ? xin_p[(b * IMG + yy) * IMG + xx] : 0.f;
      lap += s * lapk[ky * 3 + kx];
    }

  float c00 = C0[(size_t)(b * 6 + 0) * HW + r];
  float c02 = C0[(size_t)(b * 6 + 2) * HW + r];
  float c03 = C0[(size_t)(b * 6 + 3) * HW + r];
  float c04 = C0[(size_t)(b * 6 + 4) * HW + r];

  float Hv = 2.f * xin - c00 + c2[0] * lap;

  // d_out layout: outputs[0,N) H[N,2N) C[2N,8N) V[8N,9N) V_hat[9N,10N)
  out[(size_t)NPX + i] = Hv;
  float* C = out + (size_t)2 * NPX;
  C[(size_t)(b * 6 + 0) * HW + r] = xin;
  C[(size_t)(b * 6 + 1) * HW + r] = c00;
  C[(size_t)(b * 6 + 2) * HW + r] = v;
  C[(size_t)(b * 6 + 3) * HW + r] = c02;
  C[(size_t)(b * 6 + 4) * HW + r] = c03;
  C[(size_t)(b * 6 + 5) * HW + r] = c04;
  out[(size_t)8 * NPX + i] = v;

  float inv = 1.f / sf[0];
  h8 z;
#pragma unroll
  for (int k = 0; k < 8; ++k) z[k] = (_Float16)0.f;
  h8 f = z;
  f[0] = (_Float16)(v * inv);
  f[1] = (_Float16)(c02 * inv);
  f[2] = (_Float16)(c03 * inv);
  f[3] = (_Float16)(c04 * inv);
  h8* ap = (h8*)(act + (size_t)i * CH);
  ap[0] = f; ap[1] = z; ap[2] = z; ap[3] = z;
}

// ---------------------------------------------------------------------------
// Kernel 2: 3x3 conv as implicit GEMM with WMMA.
//   MODE 0: out = relu(conv(in) + bias)                  -> f16 NHWC
//   MODE 1: out = relu(conv(in) + bias + resid)          -> f16 NHWC
//   MODE 2: y = conv(in)[ch0] + bias; vh = y*sf;
//           outputs = H + vh ; V_hat = vh                -> fp32
// One workgroup (256 thr = 8 waves) computes a 32x32 output tile.
// Each wave: strips of 16 pixels; per strip, 9 taps x (2 M-halves) WMMAs K=32.
// __launch_bounds__(256,1): LDS (92KB/WG) bounds residency anyway; give the
// RA the full register budget so the 18 A fragments (144 VGPRs) never spill.
// ---------------------------------------------------------------------------
template <int MODE>
__global__ void __launch_bounds__(256, 1) conv_kernel(
                            const _Float16* __restrict__ actIn,
                            const float* __restrict__ w,
                            const float* __restrict__ bias,
                            int Cin, int Cout,
                            _Float16* __restrict__ actOut,
                            const _Float16* __restrict__ resid,
                            const float* __restrict__ Hbuf,
                            const float* __restrict__ sf,
                            float* __restrict__ outp,
                            float* __restrict__ vhat) {
  extern __shared__ _Float16 smem[];  // [HALO*HALO][PITCH]

  const int b  = blockIdx.z;
  const int y0 = blockIdx.y * TILE;
  const int x0 = blockIdx.x * TILE;
  const int tid  = threadIdx.x;
  const int lane = tid & 31;
  const int m    = lane & 15;   // M row (out-ch) / N col (pixel) within fragment
  const int hi   = lane >> 4;   // lane-half selector per ISA layouts
  const int wv   = tid >> 5;    // wave id 0..7

  // ---- stage input tile (with halo, zero-padded) into LDS ----
  for (int p = tid; p < HALO * HALO; p += 256) {
    int ly = p / HALO, lx = p % HALO;
    int gy = y0 + ly - 1, gx = x0 + lx - 1;
    h8 z;
#pragma unroll
    for (int k = 0; k < 8; ++k) z[k] = (_Float16)0.f;
    h8 a0 = z, a1 = z, a2 = z, a3 = z;
    if (gy >= 0 && gy < IMG && gx >= 0 && gx < IMG) {
      const h8* src = (const h8*)(actIn + (((size_t)(b * IMG + gy)) * IMG + gx) * CH);
      a0 = src[0]; a1 = src[1]; a2 = src[2]; a3 = src[3];
    }
    h8* dst = (h8*)(smem + (size_t)p * PITCH);
    dst[0] = a0; dst[1] = a1; dst[2] = a2; dst[3] = a3;
  }
  __syncthreads();

  // ---- preload A fragments (weights) per ISA 16-bit A 16x32 layout ----
  // lane l: row M = l&15; element j -> K = (j<8 ? hi*8+j : 16 + hi*8 + j-8)
  v16h A[9][2];
#pragma unroll
  for (int t = 0; t < 9; ++t) {
    int ky = t / 3, kx = t % 3;
#pragma unroll
    for (int s = 0; s < 2; ++s) {
      if (MODE == 2 && s == 1) break;
      int o = s * 16 + m;
      v16h av;
#pragma unroll
      for (int j = 0; j < 16; ++j) {
        int k = (j < 8) ? (hi * 8 + j) : (16 + hi * 8 + (j - 8));
        float wvf = (o < Cout && k < Cin) ? w[((o * Cin + k) * 3 + ky) * 3 + kx] : 0.f;
        av[j] = (_Float16)wvf;
      }
      A[t][s] = av;
    }
  }

  // ---- 64 strips of 16 px; wave wv takes strips wv, wv+8, ... ----
  for (int nt = wv; nt < TILE * 2; nt += 8) {
    int ly  = nt >> 1;
    int lx0 = (nt & 1) * 16;
    int px  = lx0 + m;  // local x of this lane's N column

    v8f acc0, acc1;
#pragma unroll
    for (int r = 0; r < 8; ++r) {
      int o0 = hi * 8 + r;
      acc0[r] = (o0 < Cout) ? bias[o0] : 0.f;
      int o1 = 16 + hi * 8 + r;
      acc1[r] = (o1 < Cout) ? bias[o1] : 0.f;
    }

    // B 32x16 f16: lane = N column; lanes 0-15 K=0..7,16..23, lanes 16-31 K=8..15,24..31.
    // Double-buffer the two ds_load_b128 per tap so the next fragment is in
    // flight while the current pair of WMMAs executes.
    const _Float16* srow = smem + (size_t)hi * 8;
    BU bcur;
    {
      const h8* bp = (const h8*)(srow + (size_t)(ly * HALO + px) * PITCH);
      bcur.h[0] = bp[0];
      bcur.h[1] = bp[2];
    }
#pragma unroll
    for (int t = 0; t < 9; ++t) {
      BU bnxt = bcur;
      if (t < 8) {
        int t2 = t + 1;
        int pix2 = (ly + t2 / 3) * HALO + (px + t2 % 3);
        const h8* bp = (const h8*)(srow + (size_t)pix2 * PITCH);
        bnxt.h[0] = bp[0];
        bnxt.h[1] = bp[2];
      }
      acc0 = __builtin_amdgcn_wmma_f32_16x16x32_f16(false, A[t][0], false, bcur.v,
                                                    (short)0, acc0, false, false);
      if (MODE != 2)
        acc1 = __builtin_amdgcn_wmma_f32_16x16x32_f16(false, A[t][1], false, bcur.v,
                                                      (short)0, acc1, false, false);
      bcur = bnxt;
    }

    int gy = y0 + ly, gx = x0 + px;
    if (MODE == 2) {
      // D row M=0 lives in element 0 of lanes 0..15
      if (hi == 0) {
        size_t gp = ((size_t)(b * IMG + gy)) * IMG + gx;
        float vh = acc0[0] * sf[0];
        vhat[gp] = vh;
        outp[gp] = Hbuf[gp] + vh;
      }
    } else {
      // lane owns channels [hi*8, hi*8+8) and [16+hi*8, 16+hi*8+8) at pixel px
      size_t base = (((size_t)(b * IMG + gy)) * IMG + gx) * CH;
      h8 o0, o1;
      if (MODE == 1) {
        h8 r0 = *(const h8*)(resid + base + hi * 8);
        h8 r1 = *(const h8*)(resid + base + 16 + hi * 8);
#pragma unroll
        for (int r = 0; r < 8; ++r) {
          float f0 = acc0[r] + (float)r0[r];
          float f1 = acc1[r] + (float)r1[r];
          o0[r] = (_Float16)(f0 > 0.f ? f0 : 0.f);
          o1[r] = (_Float16)(f1 > 0.f ? f1 : 0.f);
        }
      } else {
#pragma unroll
        for (int r = 0; r < 8; ++r) {
          float f0 = acc0[r];
          float f1 = acc1[r];
          o0[r] = (_Float16)(f0 > 0.f ? f0 : 0.f);
          o1[r] = (_Float16)(f1 > 0.f ? f1 : 0.f);
        }
      }
      *(h8*)(actOut + base + hi * 8)      = o0;
      *(h8*)(actOut + base + 16 + hi * 8) = o1;
    }
  }
}

// ---------------------------------------------------------------------------
// launch: prep -> conv_in -> 3x(conv,conv+res) -> conv_out(+final epilogue)
// d_in order (setup_inputs insertion order, recursively flattened):
//  0 inputs, 1 H0, 2 C0, 3 c2, 4 sf, 5 lap_kernel,
//  6 w_in, 7 b_in, 8..19 (w1,b1,w2,b2)x3, 20 w_out, 21 b_out
// ---------------------------------------------------------------------------
extern "C" void kernel_launch(void* const* d_in, const int* in_sizes, int n_in,
                              void* d_out, int out_size, void* d_ws, size_t ws_size,
                              hipStream_t stream) {
  (void)in_sizes; (void)n_in; (void)out_size; (void)ws_size;
  const float* xin  = (const float*)d_in[0];
  const float* H0   = (const float*)d_in[1];
  const float* C0   = (const float*)d_in[2];
  const float* c2   = (const float*)d_in[3];
  const float* sf   = (const float*)d_in[4];
  const float* lapk = (const float*)d_in[5];
  const float* w_in = (const float*)d_in[6];
  const float* b_in = (const float*)d_in[7];
  const float* w_out= (const float*)d_in[20];
  const float* b_out= (const float*)d_in[21];
  float* out = (float*)d_out;

  _Float16* actA = (_Float16*)d_ws;                    // 8*512*512*32 f16 = 128 MB
  _Float16* actB = actA + (size_t)NPX * CH;            // second 128 MB

  prep_kernel<<<(NPX + 255) / 256, 256, 0, stream>>>(xin, H0, C0, c2, sf, lapk,
                                                     out, actA);

  dim3 grid(IMG / TILE, IMG / TILE, NBATCH);  // 16 x 16 x 8
  // conv_in: 4 real input channels (rest zero-padded), relu
  conv_kernel<0><<<grid, 256, SMEM_BYTES, stream>>>(
      actA, w_in, b_in, 4, 32, actB, nullptr, nullptr, nullptr, nullptr, nullptr);

  for (int blk = 0; blk < 3; ++blk) {
    const float* w1 = (const float*)d_in[8 + blk * 4 + 0];
    const float* b1 = (const float*)d_in[8 + blk * 4 + 1];
    const float* w2 = (const float*)d_in[8 + blk * 4 + 2];
    const float* b2 = (const float*)d_in[8 + blk * 4 + 3];
    // r = relu(conv1(h)) : actB -> actA
    conv_kernel<0><<<grid, 256, SMEM_BYTES, stream>>>(
        actB, w1, b1, 32, 32, actA, nullptr, nullptr, nullptr, nullptr, nullptr);
    // h = relu(h + conv2(r)) : actA -> actB, residual read from actB (same px)
    conv_kernel<1><<<grid, 256, SMEM_BYTES, stream>>>(
        actA, w2, b2, 32, 32, actB, actB, nullptr, nullptr, nullptr, nullptr);
  }

  // conv_out + final epilogue: outputs = H + sf*y ; V_hat = sf*y
  conv_kernel<2><<<grid, 256, SMEM_BYTES, stream>>>(
      actB, w_out, b_out, 32, 1, nullptr, nullptr,
      out + (size_t)NPX,          // H
      sf,
      out,                        // outputs
      out + (size_t)9 * NPX);     // V_hat
}
